// ResidualBlock_35321811042894
// MI455X (gfx1250) — matrix operand order repaired and checked by reading, compiled, not run on hardware
//
#include <hip/hip_runtime.h>
#include <hip/hip_bf16.h>

// ---------------------------------------------------------------------------
// Types for CDNA5 WMMA (wave32): v_wmma_f32_16x16x32_bf16
// ---------------------------------------------------------------------------
typedef __attribute__((ext_vector_type(16))) __bf16 v16bf;
typedef __attribute__((ext_vector_type(8)))  __bf16 v8bf;
typedef __attribute__((ext_vector_type(8)))  float  v8f;
typedef __attribute__((ext_vector_type(4)))  float  f4;

__device__ __forceinline__ __bf16 f2bf(float f) {
  union { float f; unsigned u; } v; v.f = f;
  unsigned r = v.u + 0x7FFFu + ((v.u >> 16) & 1u);   // round-to-nearest-even
  union { unsigned short s; __bf16 b; } o; o.s = (unsigned short)(r >> 16);
  return o.b;
}

__device__ __forceinline__ float elu1(float v) {
  return v > 0.f ? v : (expf(v) - 1.f);
}

// ---------------------------------------------------------------------------
// Utility kernels
// ---------------------------------------------------------------------------
__global__ void zero_f32(float* __restrict__ p, long n) {
  long i = (long)blockIdx.x * blockDim.x + threadIdx.x;
  long stride = (long)gridDim.x * blockDim.x;
  for (; i < n; i += stride) p[i] = 0.f;
}

// copy x into the appended reduction columns of h1ext and hsext
__global__ void put_x(const float* __restrict__ x, float* __restrict__ h1,
                      float* __restrict__ hs, int N) {
  int i = blockIdx.x * blockDim.x + threadIdx.x;
  if (i >= N * 32) return;
  int n = i >> 5, c = i & 31;
  float v = x[i];
  h1[(long)n * 832 + 800 + c] = v;
  hs[(long)n * 64 + 32 + c]   = v;
}

// copy layer-1 activations into appended columns of h2ext
__global__ void put_h(const float* __restrict__ H, float* __restrict__ h2, int N) {
  int i = blockIdx.x * blockDim.x + threadIdx.x;
  if (i >= N * 64) return;
  int n = i >> 6, c = i & 63;
  h2[(long)n * 1664 + 1600 + c] = H[i];
}

__global__ void count_deg(const int* __restrict__ dst, float* __restrict__ deg, long E) {
  long e = (long)blockIdx.x * blockDim.x + threadIdx.x;
  if (e >= E) return;
  atomicAdd(&deg[dst[e]], 1.f);
}

__global__ void recip_deg(const float* __restrict__ deg, float* __restrict__ rdeg, int N) {
  int n = blockIdx.x * blockDim.x + threadIdx.x;
  if (n >= N) return;
  rdeg[n] = 1.f / fmaxf(deg[n], 1.f);
}

// ---------------------------------------------------------------------------
// Spline scatter: 32 lanes (= channels) per edge. Deg division folded in.
// Corner k-index: wi = (f0+s0) + 5*(f1+s1); bucket column = wi*C + c.
// ---------------------------------------------------------------------------
__global__ void scatter_c32(const float* __restrict__ x, const int* __restrict__ src,
                            const int* __restrict__ dst, const float* __restrict__ attr,
                            const float* __restrict__ rdeg,
                            float* __restrict__ h1, float* __restrict__ hs, long E) {
  long t = (long)blockIdx.x * blockDim.x + threadIdx.x;
  long e = t >> 5; int c = (int)(t & 31);
  if (e >= E) return;
  int s = src[e], d = dst[e];
  float w  = rdeg[d];
  float wx = w * x[(long)s * 32 + c];
  atomicAdd(&hs[(long)d * 64 + c], wx);              // shortcut mean-agg
  float v0 = attr[e * 2 + 0] * 4.f, v1 = attr[e * 2 + 1] * 4.f;
  float f0 = fminf(fmaxf(floorf(v0), 0.f), 3.f);
  float f1 = fminf(fmaxf(floorf(v1), 0.f), 3.f);
  float fr0 = v0 - f0, fr1 = v1 - f1;
  int i0 = (int)f0, i1 = (int)f1;
#pragma unroll
  for (int s1 = 0; s1 < 2; ++s1)
#pragma unroll
    for (int s0 = 0; s0 < 2; ++s0) {
      float b  = (s0 ? fr0 : 1.f - fr0) * (s1 ? fr1 : 1.f - fr1);
      int   wi = (i0 + s0) + 5 * (i1 + s1);
      atomicAdd(&h1[(long)d * 832 + wi * 32 + c], b * wx);
    }
}

// same but 64 channels (layer 2), row stride 1664
__global__ void scatter_c64(const float* __restrict__ H, const int* __restrict__ src,
                            const int* __restrict__ dst, const float* __restrict__ attr,
                            const float* __restrict__ rdeg,
                            float* __restrict__ h2, long E) {
  long t = (long)blockIdx.x * blockDim.x + threadIdx.x;
  long e = t >> 6; int c = (int)(t & 63);
  if (e >= E) return;
  int s = src[e], d = dst[e];
  float wx = rdeg[d] * H[(long)s * 64 + c];
  float v0 = attr[e * 2 + 0] * 4.f, v1 = attr[e * 2 + 1] * 4.f;
  float f0 = fminf(fmaxf(floorf(v0), 0.f), 3.f);
  float f1 = fminf(fmaxf(floorf(v1), 0.f), 3.f);
  float fr0 = v0 - f0, fr1 = v1 - f1;
  int i0 = (int)f0, i1 = (int)f1;
#pragma unroll
  for (int s1 = 0; s1 < 2; ++s1)
#pragma unroll
    for (int s0 = 0; s0 < 2; ++s0) {
      float b  = (s0 ? fr0 : 1.f - fr0) * (s1 ? fr1 : 1.f - fr1);
      int   wi = (i0 + s0) + 5 * (i1 + s1);
      atomicAdd(&h2[(long)d * 1664 + wi * 64 + c], b * wx);
    }
}

// ---------------------------------------------------------------------------
// WMMA GEMM: Y[N x 64] = A[N x Kdim] @ W + bias.
// W rows < K0 from W0, rows >= K0 from W1 (appended root weight).
// Whole weight matrix is staged in LDS in B-fragment layout (<=208KB, fits
// CDNA5's 320KB LDS). One wave computes a 32x64 output tile: two A fragments
// share the four B fragments -> 8 WMMAs per K-step, 1 ds_load_b128 per WMMA.
// ---------------------------------------------------------------------------
__device__ __forceinline__ v16bf ld_frag(const __bf16* p) {
  v8bf lo = *(const v8bf*)p;
  v8bf hi = *(const v8bf*)(p + 8);
  v16bf r;
#pragma unroll
  for (int i = 0; i < 8; ++i) { r[i] = lo[i]; r[8 + i] = hi[i]; }
  return r;
}

__device__ __forceinline__ v16bf mk_afrag(const float* arow, int koff) {
  // lane holds row M, K = kb..kb+3, kb+4..kb+7, kb+16..19, kb+20..23
  const f4* q0 = (const f4*)(arow + koff);
  const f4* q1 = (const f4*)(arow + koff + 16);
  f4 u0 = q0[0], u1 = q0[1], u2 = q1[0], u3 = q1[1];
  v16bf a;
#pragma unroll
  for (int i = 0; i < 4; ++i) {
    a[i]      = f2bf(u0[i]);
    a[4 + i]  = f2bf(u1[i]);
    a[8 + i]  = f2bf(u2[i]);
    a[12 + i] = f2bf(u3[i]);
  }
  return a;
}

__global__ void gemm_bn(const float* __restrict__ A, int Kdim, int K0,
                        const float* __restrict__ W0, const float* __restrict__ W1,
                        const float* __restrict__ bias,
                        float* __restrict__ Y, int N) {
  extern __shared__ __bf16 sW[];   // Kdim*64 bf16, fragment-ordered
  const int tid = threadIdx.x;

  // Fill LDS in exact B-fragment layout (ISA 7.12.2, 16-bit B 32x16):
  // frag(kt,nt): lane L holds col nt*16+(L&15), K rows kt*32+(L>=16?8:0)+(j&7)+(j>=8?16:0).
  const int total = Kdim * 64;
  for (int idx = tid; idx < total; idx += blockDim.x) {
    const int j  = idx & 15;
    const int ln = (idx >> 4) & 31;
    const int nt = (idx >> 9) & 3;
    const int kt = idx >> 11;
    const int r  = kt * 32 + ((ln >> 4) << 3) + (j & 7) + ((j >> 3) << 4);
    const int c  = (nt << 4) + (ln & 15);
    const float v = (r < K0) ? W0[r * 64 + c] : W1[(r - K0) * 64 + c];
    sW[idx] = f2bf(v);
  }
  __syncthreads();

  const int lane  = tid & 31;
  const int wave  = tid >> 5;
  const int mbase = (blockIdx.x * (blockDim.x >> 5) + wave) << 5;  // 32 rows/wave
  int row0 = mbase + (lane & 15);
  int row1 = mbase + 16 + (lane & 15);
  if (row0 >= N) row0 = N - 1;                   // clamp: keep EXEC all-ones
  if (row1 >= N) row1 = N - 1;
  const int kb = (lane >> 4) << 3;               // 0 or 8 (A-matrix K half)
  const float* arow0 = A + (long)row0 * Kdim;
  const float* arow1 = A + (long)row1 * Kdim;

  v8f acc00 = {}; v8f acc01 = {}; v8f acc02 = {}; v8f acc03 = {};
  v8f acc10 = {}; v8f acc11 = {}; v8f acc12 = {}; v8f acc13 = {};
  const int nkt = Kdim >> 5;
  for (int kt = 0; kt < nkt; ++kt) {
    const int koff = (kt << 5) + kb;
    v16bf a0 = mk_afrag(arow0, koff);
    v16bf a1 = mk_afrag(arow1, koff);
    const int fb = (kt << 2) * 32 + lane;        // fragment base (v16bf units)
    v16bf b0 = ld_frag(sW + ((long)(fb +  0) << 4));
    v16bf b1 = ld_frag(sW + ((long)(fb + 32) << 4));
    v16bf b2 = ld_frag(sW + ((long)(fb + 64) << 4));
    v16bf b3 = ld_frag(sW + ((long)(fb + 96) << 4));
    acc00 = __builtin_amdgcn_wmma_f32_16x16x32_bf16(false, a0, false, b0, (short)0, acc00, false, false);
    acc10 = __builtin_amdgcn_wmma_f32_16x16x32_bf16(false, a1, false, b0, (short)0, acc10, false, false);
    acc01 = __builtin_amdgcn_wmma_f32_16x16x32_bf16(false, a0, false, b1, (short)0, acc01, false, false);
    acc11 = __builtin_amdgcn_wmma_f32_16x16x32_bf16(false, a1, false, b1, (short)0, acc11, false, false);
    acc02 = __builtin_amdgcn_wmma_f32_16x16x32_bf16(false, a0, false, b2, (short)0, acc02, false, false);
    acc12 = __builtin_amdgcn_wmma_f32_16x16x32_bf16(false, a1, false, b2, (short)0, acc12, false, false);
    acc03 = __builtin_amdgcn_wmma_f32_16x16x32_bf16(false, a0, false, b3, (short)0, acc03, false, false);
    acc13 = __builtin_amdgcn_wmma_f32_16x16x32_bf16(false, a1, false, b3, (short)0, acc13, false, false);
  }

  // C/D layout: VGPR i -> row (lane>=16 ? 8 : 0) + i, col (lane&15)+nt*16
  const int rb0   = mbase + ((lane >> 4) << 3);
  const int cbase = lane & 15;
  const float bia0 = bias[cbase +  0];
  const float bia1 = bias[cbase + 16];
  const float bia2 = bias[cbase + 32];
  const float bia3 = bias[cbase + 48];
#pragma unroll
  for (int i = 0; i < 8; ++i) {
    int r = rb0 + i;
    if (r < N) {
      long o = (long)r * 64 + cbase;
      Y[o +  0] = acc00[i] + bia0;
      Y[o + 16] = acc01[i] + bia1;
      Y[o + 32] = acc02[i] + bia2;
      Y[o + 48] = acc03[i] + bia3;
    }
    int r2 = r + 16;
    if (r2 < N) {
      long o = (long)r2 * 64 + cbase;
      Y[o +  0] = acc10[i] + bia0;
      Y[o + 16] = acc11[i] + bia1;
      Y[o + 32] = acc12[i] + bia2;
      Y[o + 48] = acc13[i] + bia3;
    }
  }
}

// ---------------------------------------------------------------------------
// Column stats (sum, sumsq over N rows of a [N,64] matrix) for batch-norm
// ---------------------------------------------------------------------------
__global__ void colstats(const float* __restrict__ Y, int N,
                         float* __restrict__ sum, float* __restrict__ sumsq) {
  __shared__ float s1[256], s2[256];
  int tid = threadIdx.x;
  int col = tid & 63, grp = tid >> 6;
  float a = 0.f, a2 = 0.f;
  for (int r = blockIdx.x * 4 + grp; r < N; r += gridDim.x * 4) {
    float v = Y[(long)r * 64 + col];
    a += v; a2 += v * v;
  }
  s1[tid] = a; s2[tid] = a2;
  __syncthreads();
  if (tid < 128) { s1[tid] += s1[tid + 128]; s2[tid] += s2[tid + 128]; }
  __syncthreads();
  if (tid < 64) {
    atomicAdd(&sum[col],   s1[tid] + s1[tid + 64]);
    atomicAdd(&sumsq[col], s2[tid] + s2[tid + 64]);
  }
}

__global__ void bn_coef(const float* __restrict__ sum, const float* __restrict__ sumsq,
                        const float* __restrict__ gamma, const float* __restrict__ beta,
                        float Nf, float* __restrict__ ab) {
  int c = threadIdx.x;
  if (c >= 64) return;
  float m   = sum[c] / Nf;
  float var = fmaxf(sumsq[c] / Nf - m * m, 0.f);
  float s   = gamma[c] * rsqrtf(var + 1e-5f);
  ab[c]      = s;
  ab[64 + c] = beta[c] - m * s;
}

__global__ void bn_elu(const float* __restrict__ Y, const float* __restrict__ ab,
                       float* __restrict__ H, long total) {
  long i = (long)blockIdx.x * blockDim.x + threadIdx.x;
  if (i >= total) return;
  int c = (int)(i & 63);
  H[i] = elu1(Y[i] * ab[c] + ab[64 + c]);
}

__global__ void final_out(const float* __restrict__ Y2, const float* __restrict__ ab2,
                          const float* __restrict__ Ys, const float* __restrict__ abS,
                          float* __restrict__ out, long total) {
  long i = (long)blockIdx.x * blockDim.x + threadIdx.x;
  if (i >= total) return;
  int c = (int)(i & 63);
  float v = Y2[i] * ab2[c] + ab2[64 + c] + Ys[i] * abS[c] + abS[64 + c];
  out[i] = elu1(v);
}

// ---------------------------------------------------------------------------
// Host-side orchestration
// ---------------------------------------------------------------------------
extern "C" void kernel_launch(void* const* d_in, const int* in_sizes, int n_in,
                              void* d_out, int out_size, void* d_ws, size_t ws_size,
                              hipStream_t stream) {
  (void)n_in; (void)out_size; (void)ws_size;
  const float* x    = (const float*)d_in[0];
  const int*   ei   = (const int*)  d_in[1];
  const float* attr = (const float*)d_in[2];
  const float* w1   = (const float*)d_in[3];
  const float* r1   = (const float*)d_in[4];
  const float* b1   = (const float*)d_in[5];
  const float* g1   = (const float*)d_in[6];
  const float* be1  = (const float*)d_in[7];
  const float* w2   = (const float*)d_in[8];
  const float* r2   = (const float*)d_in[9];
  const float* b2   = (const float*)d_in[10];
  const float* g2   = (const float*)d_in[11];
  const float* be2  = (const float*)d_in[12];
  const float* wS   = (const float*)d_in[13];
  const float* rS   = (const float*)d_in[14];
  const float* bS   = (const float*)d_in[15];
  const float* gS   = (const float*)d_in[16];
  const float* beS  = (const float*)d_in[17];

  const int  N = in_sizes[0] / 32;
  const long E = in_sizes[1] / 2;
  const int* srcp = ei;
  const int* dstp = ei + E;
  float* outp = (float*)d_out;

  // workspace layout (hbig is reused: h1ext first, then h2ext)
  float* wsf = (float*)d_ws;
  long off = 0;
  float* hbig = wsf + off; off += (long)N * 1664;
  float* hs   = wsf + off; off += (long)N * 64;
  float* y1   = wsf + off; off += (long)N * 64;
  float* y2   = wsf + off; off += (long)N * 64;
  float* ysh  = wsf + off; off += (long)N * 64;
  float* H    = wsf + off; off += (long)N * 64;
  float* deg  = wsf + off; off += N;
  float* rdeg = wsf + off; off += N;
  float* st   = wsf + off; off += 6 * 64;   // sum1,sq1,sum2,sq2,sumS,sqS
  float* ab   = wsf + off; off += 3 * 128;  // ab1, ab2, abS
  float* sum1 = st, *sq1 = st + 64, *sum2 = st + 128, *sq2 = st + 192,
       * sumS = st + 256, *sqS = st + 320;
  float* ab1 = ab, *ab2 = ab + 128, *abS = ab + 256;

  const long NE32 = (long)N * 32, NE64 = (long)N * 64;
  const int  gM   = ((N + 31) / 32 + 7) / 8;   // 32 rows/wave, 8 waves/block

  // phase 0: zero accumulation buffers
  zero_f32<<<2048, 256, 0, stream>>>(hbig, (long)N * 832);
  zero_f32<<<2048, 256, 0, stream>>>(hs, NE64);
  zero_f32<<<256, 256, 0, stream>>>(deg, N);
  zero_f32<<<1, 256, 0, stream>>>(st, 6 * 64);
  put_x<<<(unsigned)((NE32 + 255) / 256), 256, 0, stream>>>(x, hbig, hs, N);

  // phase 1: degree
  count_deg<<<(unsigned)((E + 255) / 256), 256, 0, stream>>>(dstp, deg, E);
  recip_deg<<<(unsigned)((N + 255) / 256), 256, 0, stream>>>(deg, rdeg, N);

  // phase 2: spline scatter for conv1 + shortcut
  scatter_c32<<<(unsigned)((E * 32 + 255) / 256), 256, 0, stream>>>(
      x, srcp, dstp, attr, rdeg, hbig, hs, E);

  // phase 3: conv1 GEMM (WMMA, K=832) -> y1 ; BN coeffs ; ELU -> H
  gemm_bn<<<gM, 256, 832 * 128, stream>>>(hbig, 832, 800, w1, r1, b1, y1, N);
  colstats<<<128, 256, 0, stream>>>(y1, N, sum1, sq1);
  bn_coef<<<1, 64, 0, stream>>>(sum1, sq1, g1, be1, (float)N, ab1);
  bn_elu<<<(unsigned)((NE64 + 255) / 256), 256, 0, stream>>>(y1, ab1, H, NE64);

  // phase 4: conv2 (reuse hbig as h2ext)
  zero_f32<<<4096, 256, 0, stream>>>(hbig, (long)N * 1664);
  put_h<<<(unsigned)((NE64 + 255) / 256), 256, 0, stream>>>(H, hbig, N);
  scatter_c64<<<(unsigned)((E * 64 + 255) / 256), 256, 0, stream>>>(
      H, srcp, dstp, attr, rdeg, hbig, E);
  gemm_bn<<<gM, 256, 1664 * 128, stream>>>(hbig, 1664, 1600, w2, r2, b2, y2, N);

  // phase 5: shortcut GEMM (K=64)
  gemm_bn<<<gM, 256, 64 * 128, stream>>>(hs, 64, 32, wS, rS, bS, ysh, N);

  // phase 6: BN for both branches, fused residual + ELU
  colstats<<<128, 256, 0, stream>>>(y2, N, sum2, sq2);
  colstats<<<128, 256, 0, stream>>>(ysh, N, sumS, sqS);
  bn_coef<<<1, 64, 0, stream>>>(sum2, sq2, g2, be2, (float)N, ab2);
  bn_coef<<<1, 64, 0, stream>>>(sumS, sqS, gS, beS, (float)N, abS);
  final_out<<<(unsigned)((NE64 + 255) / 256), 256, 0, stream>>>(
      y2, ab2, ysh, abS, outp, NE64);
}
